// DeepScaffold_16793322127441
// MI455X (gfx1250) — matrix operand order repaired and checked by reading, not compiled, not run on hardware
//
#include <hip/hip_runtime.h>
#include <hip/hip_bf16.h>
#include <math.h>

typedef __attribute__((ext_vector_type(16))) _Float16 v16h;
typedef __attribute__((ext_vector_type(8)))  _Float16 v8h;
typedef __attribute__((ext_vector_type(4)))  _Float16 h4;
typedef __attribute__((ext_vector_type(8)))  float    v8f;
typedef __attribute__((ext_vector_type(4)))  float    f4;

#define N_ATOMS  50000
#define N_BONDS  150000
#define N_BLOCKS 1000
#define NAT      40
#define NBT      7
#define EMBF     64
#define KCH      32
#define L_LAYERS 10
#define OUTF     256
#define NLOGIT   164

// ---------------- utility kernels ----------------

__global__ void k_zero(float* p, long n) {
  long i = (long)blockIdx.x * 256 + threadIdx.x;
  if (i < n) p[i] = 0.f;
}

__global__ void k_fill(float* p, long n, float v) {
  long i = (long)blockIdx.x * 256 + threadIdx.x;
  if (i < n) p[i] = v;
}

__device__ inline void atomicMaxF(float* addr, float v) {
  unsigned* u = (unsigned*)addr;
  unsigned old = *u;
  while (true) {
    float f = __uint_as_float(old);
    if (f >= v) break;
    unsigned assumed = old;
    old = atomicCAS(u, assumed, __float_as_uint(v));
    if (old == assumed) break;
  }
}

// ---------------- embedding ----------------

__global__ void k_embed(const int* __restrict__ at, const int* __restrict__ isc,
                        const int* __restrict__ lam, const float* __restrict__ emb,
                        float* __restrict__ feats) {
  long i = (long)blockIdx.x * 256 + threadIdx.x;
  if (i >= (long)N_ATOMS * EMBF) return;
  int atom = (int)(i / EMBF), f = (int)(i % EMBF);
  int t = at[atom];
  if (isc[atom] == 1)       t += 2 * NAT;   // +NAT twice (reference quirk)
  else if (lam[atom] == 1)  t += 2 * NAT;
  else if (lam[atom] == 2)  t += 3 * NAT;
  feats[(long)atom * 384 + f] = emb[(long)t * EMBF + f];
}

// ---------------- per-column BN folded to scale/shift ----------------
// scale = rsqrt(var+eps)*g ; shift = b - mean*scale

__global__ void k_colstats(const float* __restrict__ X, int ld, int M,
                           const float* __restrict__ g, const float* __restrict__ b,
                           float* __restrict__ S, float* __restrict__ T) {
  int col = blockIdx.x;
  float s = 0.f, s2 = 0.f;
  for (int r = threadIdx.x; r < M; r += 256) {
    float v = X[(long)r * ld + col];
    s += v; s2 += v * v;
  }
  __shared__ float sh[256], sh2[256];
  sh[threadIdx.x] = s; sh2[threadIdx.x] = s2;
  __syncthreads();
  for (int o = 128; o > 0; o >>= 1) {
    if (threadIdx.x < o) { sh[threadIdx.x] += sh[threadIdx.x + o]; sh2[threadIdx.x] += sh2[threadIdx.x + o]; }
    __syncthreads();
  }
  if (threadIdx.x == 0) {
    float m = sh[0] / (float)M;
    float var = sh2[0] / (float)M - m * m;
    float sc = rsqrtf(var + 1e-5f) * g[col];
    S[col] = sc;
    T[col] = b[col] - m * sc;
  }
}

// ---------------- weight fp32 -> fp16 transposed [Npad][K], zero-padded ----------------

__global__ void k_convw(const float* __restrict__ W, int K, int Nc, int Npad,
                        _Float16* __restrict__ WHT) {
  long i = (long)blockIdx.x * 256 + threadIdx.x;
  if (i >= (long)K * Npad) return;
  int n = (int)(i / K), k = (int)(i % K);
  WHT[i] = (n < Nc) ? (_Float16)W[(long)k * Nc + n] : (_Float16)0.f;
}

// ---------------- fused BN+ELU+GEMM via WMMA ----------------
// C[M,Nc] = elu(A*S + T) @ W + cbias. 64x64 tile/block, 8 waves (4x2),
// each wave reuses its A fragment against two B fragments (two WMMAs).
// LDS tiles staged directly in wave32 WMMA fragment order -> ds_load_b128 reads.
// Requires K%32==0, Npad%64==0.

__global__ __launch_bounds__(256) void k_gemm_bnrl(
    const float* __restrict__ A, int lda, int M, int K,
    const float* __restrict__ S, const float* __restrict__ T,
    const _Float16* __restrict__ WHT,          // [Npad][K] column-major weights
    const float* __restrict__ cbias, int Nc,
    float* __restrict__ C, int ldc) {
  __shared__ _Float16 AF[4][32][24];   // [mTile][lane][16 used, pad->24]
  __shared__ _Float16 BF[4][32][24];   // [nTile][lane][16 used]
  int tid = threadIdx.x;
  int lane = tid & 31, wid = tid >> 5;
  int wm = wid & 3, wn = wid >> 2;     // 4x2 waves; wave covers 16 rows x 32 cols
  int gg = lane >> 4, lm = lane & 15;
  int m0 = blockIdx.x * 64;
  int n0 = blockIdx.y * 64;
  v8f acc0 = {}, acc1 = {};

  for (int k0 = 0; k0 < K; k0 += 32) {
    // ---- A tile: 64 rows x 32 k, float4 loads, BN+ELU, packed b64 LDS stores
    #pragma unroll
    for (int t = 0; t < 2; t++) {
      int q  = tid + t * 256;          // 0..511 quads
      int r  = q >> 3;                 // row 0..63
      int kk = (q & 7) << 2;           // k offset 0,4,..,28
      int gr = m0 + r, gk = k0 + kk;
      f4 x = {0.f, 0.f, 0.f, 0.f};
      if (gr < M) x = *(const f4*)(A + (long)gr * lda + gk);
      f4 sc = *(const f4*)(S + gk);
      f4 sh = *(const f4*)(T + gk);
      h4 hv;
      #pragma unroll
      for (int e = 0; e < 4; e++) {
        float xn = x[e] * sc[e] + sh[e];
        xn = (xn > 0.f) ? xn : (__expf(xn) - 1.f);
        hv[e] = (_Float16)xn;
      }
      int fg = (kk >> 3) & 1;                       // lane half
      int j  = ((kk >> 4) << 3) | (kk & 7);         // fragment element (j..j+3)
      *(h4*)(&AF[r >> 4][(fg << 4) | (r & 15)][j]) = hv;
    }
    // ---- B tile: 32 k x 64 n from transposed f16 weights, b64 LDS stores
    #pragma unroll
    for (int t = 0; t < 2; t++) {
      int q  = tid + t * 256;          // 0..511 quads
      int n  = q >> 3;                 // 0..63
      int kk = (q & 7) << 2;           // 0,4,..,28
      h4 hv = *(const h4*)(WHT + (long)(n0 + n) * K + k0 + kk);
      int fg = (kk >> 3) & 1;                       // B: same k order as A
      int j  = ((kk >> 4) << 3) | (kk & 7);
      *(h4*)(&BF[n >> 4][(fg << 4) | (n & 15)][j]) = hv;
    }
    __syncthreads();

    // ---- fragments are contiguous per lane: b128 LDS loads
    v8h a0 = *(const v8h*)(&AF[wm][lane][0]);
    v8h a1 = *(const v8h*)(&AF[wm][lane][8]);
    v16h af = __builtin_shufflevector(a0, a1, 0,1,2,3,4,5,6,7,8,9,10,11,12,13,14,15);
    v8h b00 = *(const v8h*)(&BF[wn * 2 + 0][lane][0]);
    v8h b01 = *(const v8h*)(&BF[wn * 2 + 0][lane][8]);
    v8h b10 = *(const v8h*)(&BF[wn * 2 + 1][lane][0]);
    v8h b11 = *(const v8h*)(&BF[wn * 2 + 1][lane][8]);
    v16h bf0 = __builtin_shufflevector(b00, b01, 0,1,2,3,4,5,6,7,8,9,10,11,12,13,14,15);
    v16h bf1 = __builtin_shufflevector(b10, b11, 0,1,2,3,4,5,6,7,8,9,10,11,12,13,14,15);
    acc0 = __builtin_amdgcn_wmma_f32_16x16x32_f16(
        false, af, false, bf0, (short)0, acc0, false, false);
    acc1 = __builtin_amdgcn_wmma_f32_16x16x32_f16(
        false, af, false, bf1, (short)0, acc1, false, false);
    __syncthreads();
  }

  int row_base = m0 + wm * 16 + (gg << 3);        // D: M = r + 8*laneHalf
  #pragma unroll
  for (int r = 0; r < 8; r++) {
    int row = row_base + r;
    if (row >= M) break;
    int col0 = n0 + wn * 32 + lm;
    int col1 = col0 + 16;
    if (col0 < Nc) C[(long)row * ldc + col0] = acc0[r] + cbias[col0];
    if (col1 < Nc) C[(long)row * ldc + col1] = acc1[r] + cbias[col1];
  }
}

// ---------------- message passing: msgs = hh[end, bt]; segsum by begin ----------------

__global__ void k_scatter(const float* __restrict__ HHp, int ldhh, int fout,
                          const int* __restrict__ bonds,
                          float* __restrict__ dst, int ldd, int dstoff) {
  long i = (long)blockIdx.x * 256 + threadIdx.x;
  if (i >= (long)N_BONDS * fout) return;
  int bnd = (int)(i / fout), f = (int)(i % fout);
  int bg = bonds[3 * bnd + 0];
  int en = bonds[3 * bnd + 1];
  int bt = bonds[3 * bnd + 2];
  atomicAdd(&dst[(long)bg * ldd + dstoff + f], HHp[(long)en * ldhh + bt * fout + f]);
}

// ---------------- block pooling / concat ----------------

__global__ void k_pool_sum(const float* __restrict__ Ap, const int* __restrict__ blk,
                           float* __restrict__ mol, float* __restrict__ cnt) {
  long i = (long)blockIdx.x * 256 + threadIdx.x;
  if (i >= (long)N_ATOMS * OUTF) return;
  int atom = (int)(i / OUTF), f = (int)(i % OUTF);
  int bkt = blk[atom];
  atomicAdd(&mol[(long)bkt * OUTF + f], Ap[i]);
  if (f == 0) atomicAdd(&cnt[bkt], 1.f);
}

__global__ void k_mol_div(float* __restrict__ mol, const float* __restrict__ cnt) {
  long i = (long)blockIdx.x * 256 + threadIdx.x;
  if (i >= (long)N_BLOCKS * OUTF) return;
  mol[i] /= fmaxf(cnt[i / OUTF], 1.f);
}

__global__ void k_cat(const float* __restrict__ OUTp, const float* __restrict__ mol,
                      const int* __restrict__ blk, float* __restrict__ CATp) {
  long i = (long)blockIdx.x * 256 + threadIdx.x;
  if (i >= (long)N_ATOMS * 512) return;
  int atom = (int)(i / 512), c = (int)(i % 512);
  CATp[i] = (c < 256) ? OUTp[(long)atom * 256 + c]
                      : mol[(long)blk[atom] * 256 + (c - 256)];
}

// ---------------- segment softmax ----------------

__global__ void k_segmax(const float* __restrict__ lg, const int* __restrict__ blk,
                         float* __restrict__ segmax) {
  int atom = blockIdx.x * 256 + threadIdx.x;
  if (atom >= N_ATOMS) return;
  const float* row = lg + (long)atom * NLOGIT;
  float m = -3.0e38f;
  for (int c = 0; c < NLOGIT; c++) m = fmaxf(m, row[c]);
  atomicMaxF(&segmax[blk[atom]], m);
}

__global__ void k_m_eb(const float* __restrict__ segmax, const float* __restrict__ bias,
                       float* __restrict__ m, float* __restrict__ eb, float* __restrict__ den) {
  int bkt = blockIdx.x * 256 + threadIdx.x;
  if (bkt >= N_BLOCKS) return;
  float mm = fmaxf(segmax[bkt], bias[bkt]);
  m[bkt] = mm;
  float e = __expf(bias[bkt] - mm);
  eb[bkt] = e;
  den[bkt] = e;   // den seeded with eb; atoms atomically add the rest
}

__global__ void k_expsum(float* __restrict__ lg, const float* __restrict__ m,
                         const int* __restrict__ blk, float* __restrict__ den) {
  int atom = blockIdx.x * 256 + threadIdx.x;
  if (atom >= N_ATOMS) return;
  float mm = m[blk[atom]];
  float s = 0.f;
  float* row = lg + (long)atom * NLOGIT;
  for (int c = 0; c < NLOGIT; c++) {
    float e = __expf(row[c] - mm);
    row[c] = e;        // logits overwritten with ex
    s += e;
  }
  atomicAdd(&den[blk[atom]], s);
}

__global__ void k_write_p(const float* __restrict__ ex, const float* __restrict__ den,
                          const int* __restrict__ blk, float* __restrict__ out) {
  long i = (long)blockIdx.x * 256 + threadIdx.x;
  if (i >= (long)N_ATOMS * NLOGIT) return;
  int atom = (int)(i / NLOGIT), col = (int)(i % NLOGIT);
  float p = ex[i] / den[blk[atom]];
  if (col < NAT * 4) out[(long)atom * (NAT * 4) + col] = p;                       // p_append
  else out[(long)N_ATOMS * (NAT * 4) + (long)atom * 4 + (col - NAT * 4)] = p;     // p_connect
}

__global__ void k_write_pend(const float* __restrict__ eb, const float* __restrict__ den,
                             float* __restrict__ out) {
  int bkt = blockIdx.x * 256 + threadIdx.x;
  if (bkt >= N_BLOCKS) return;
  out[(long)N_ATOMS * NLOGIT + bkt] = eb[bkt] / den[bkt];
}

// ---------------- host-side fused bnrl->GEMM helper ----------------

static void bnrl_gemm(hipStream_t s, const float* X, int lda, int M, int K,
                      const float* g, const float* b, const float* W, const float* c,
                      int Nc, int Npad, float* Y, int ldc,
                      float* Sbuf, float* Tbuf, _Float16* WHT) {
  k_colstats<<<dim3(K), dim3(256), 0, s>>>(X, lda, M, g, b, Sbuf, Tbuf);
  long wn = (long)K * Npad;
  k_convw<<<dim3((unsigned)((wn + 255) / 256)), dim3(256), 0, s>>>(W, K, Nc, Npad, WHT);
  dim3 grid((M + 63) / 64, Npad / 64);
  k_gemm_bnrl<<<grid, dim3(256), 0, s>>>(X, lda, M, K, Sbuf, Tbuf, WHT, c, Nc, Y, ldc);
}

static inline unsigned gdiv(long n) { return (unsigned)((n + 255) / 256); }

extern "C" void kernel_launch(void* const* d_in, const int* in_sizes, int n_in,
                              void* d_out, int out_size, void* d_ws, size_t ws_size,
                              hipStream_t stream) {
  (void)in_sizes; (void)n_in; (void)out_size; (void)ws_size;

  // --- inputs ---
  const int* atom_types = (const int*)d_in[0];
  const int* is_scaffold = (const int*)d_in[1];
  const int* bond_info = (const int*)d_in[2];
  const int* block_ids = (const int*)d_in[3];
  const int* last_append = (const int*)d_in[4];

  // params flatten (JAX dict alphabetical: ac, emb, end, final, layers, pool; lin: W,b,c,g)
  const float* acW   = (const float*)d_in[5];
  const float* acB   = (const float*)d_in[6];
  const float* acC   = (const float*)d_in[7];
  const float* acG   = (const float*)d_in[8];
  const float* emb   = (const float*)d_in[9];
  const float* endW  = (const float*)d_in[10];
  const float* endB  = (const float*)d_in[11];
  const float* endC  = (const float*)d_in[12];
  const float* endG  = (const float*)d_in[13];
  const float* finW  = (const float*)d_in[14];
  const float* finB  = (const float*)d_in[15];
  const float* finC  = (const float*)d_in[16];
  const float* finG  = (const float*)d_in[17];
  const int LBASE = 18;                    // 16 leaves per layer, 10 layers
  const float* poolW = (const float*)d_in[LBASE + 160 + 0];
  const float* poolB = (const float*)d_in[LBASE + 160 + 1];
  const float* poolC = (const float*)d_in[LBASE + 160 + 2];
  const float* poolG = (const float*)d_in[LBASE + 160 + 3];

  float* out = (float*)d_out;

  // --- workspace carve-up ---
  float* w = (float*)d_ws;
  float* FEATS = w;                 w += (long)N_ATOMS * 384;   // feats (ld 384)
  float* HH    = w;                 w += (long)N_ATOMS * 448;   // conv GEMM out / reused as LOGITS
  float* H0    = w;                 w += (long)N_ATOMS * 64;
  float* H1    = w;                 w += (long)N_ATOMS * 64;
  float* OUTB  = w;                 w += (long)N_ATOMS * 256;
  float* AB    = w;                 w += (long)N_ATOMS * 256;
  float* CAT   = w;                 w += (long)N_ATOMS * 512;
  float* MOL   = w;                 w += (long)N_BLOCKS * 256;
  float* CNT   = w;                 w += N_BLOCKS;
  float* BIAS  = w;                 w += N_BLOCKS;
  float* SEGMX = w;                 w += N_BLOCKS;
  float* MV    = w;                 w += N_BLOCKS;
  float* EBV   = w;                 w += N_BLOCKS;
  float* DEN   = w;                 w += N_BLOCKS;
  float* SB    = w;                 w += 512;   // folded BN scale
  float* TB    = w;                 w += 512;   // folded BN shift
  _Float16* WHT = (_Float16*)w;     // transposed f16 weights, <= 512*448 halves

  float* LOGITS = HH;               // alias: HH is free after conv layers

  // --- embedding into zeroed FEATS ---
  k_zero<<<gdiv((long)N_ATOMS * 384), 256, 0, stream>>>(FEATS, (long)N_ATOMS * 384);
  k_embed<<<gdiv((long)N_ATOMS * EMBF), 256, 0, stream>>>(atom_types, is_scaffold, last_append, emb, FEATS);

  // --- 10 graph-conv layers ---
  for (int i = 0; i < L_LAYERS; i++) {
    int base = LBASE + 16 * i;
    int din = EMBF + KCH * i;
    const float* bnW = (const float*)d_in[base + 0];
    const float* bnB = (const float*)d_in[base + 1];
    const float* bnC = (const float*)d_in[base + 2];
    const float* bnG = (const float*)d_in[base + 3];

    // h = bnrl(bn, feats) -> [N,64]
    bnrl_gemm(stream, FEATS, 384, N_ATOMS, din, bnG, bnB, bnW, bnC, 64, 64, H0, 64, SB, TB, WHT);

    float* hin = H0; float* hout = H1;
    for (int j = 0; j < 3; j++) {
      int fout = (j == 2) ? KCH : 64;
      int Nc = fout * NBT;                 // 448 / 448 / 224
      int Npad = (j == 2) ? 256 : 448;     // multiples of 64
      const float* cW = (const float*)d_in[base + 4 + 4 * j];
      const float* cB = (const float*)d_in[base + 5 + 4 * j];
      const float* cC = (const float*)d_in[base + 6 + 4 * j];
      const float* cG = (const float*)d_in[base + 7 + 4 * j];

      // hh = bnrl(conv_j, h) -> [N, 7*fout]
      bnrl_gemm(stream, hin, 64, N_ATOMS, 64, cG, cB, cW, cC, Nc, Npad, HH, Nc, SB, TB, WHT);

      if (j < 2) {
        k_zero<<<gdiv((long)N_ATOMS * 64), 256, 0, stream>>>(hout, (long)N_ATOMS * 64);
        k_scatter<<<gdiv((long)N_BONDS * fout), 256, 0, stream>>>(HH, Nc, fout, bond_info, hout, 64, 0);
        float* t = hin; hin = hout; hout = t;
      } else {
        // final conv of layer writes straight into the feats concat slice (pre-zeroed)
        k_scatter<<<gdiv((long)N_BONDS * fout), 256, 0, stream>>>(HH, Nc, fout, bond_info,
                                                                  FEATS, 384, EMBF + KCH * i);
      }
    }
  }

  // --- head ---
  bnrl_gemm(stream, FEATS, 384, N_ATOMS, 384, finG, finB, finW, finC, 256, 256, OUTB, 256, SB, TB, WHT);
  bnrl_gemm(stream, OUTB, 256, N_ATOMS, 256, poolG, poolB, poolW, poolC, 256, 256, AB, 256, SB, TB, WHT);

  k_zero<<<gdiv((long)N_BLOCKS * 256), 256, 0, stream>>>(MOL, (long)N_BLOCKS * 256);
  k_zero<<<gdiv(N_BLOCKS), 256, 0, stream>>>(CNT, N_BLOCKS);
  k_pool_sum<<<gdiv((long)N_ATOMS * 256), 256, 0, stream>>>(AB, block_ids, MOL, CNT);
  k_mol_div<<<gdiv((long)N_BLOCKS * 256), 256, 0, stream>>>(MOL, CNT);
  k_cat<<<gdiv((long)N_ATOMS * 512), 256, 0, stream>>>(OUTB, MOL, block_ids, CAT);

  // logits = bnrl(ac, cat) -> [N,164] (pad N to 192)
  bnrl_gemm(stream, CAT, 512, N_ATOMS, 512, acG, acB, acW, acC, NLOGIT, 192, LOGITS, NLOGIT, SB, TB, WHT);
  // bias = bnrl(end, mol)[:,0] -> [1000] (pad N to 64)
  bnrl_gemm(stream, MOL, 256, N_BLOCKS, 256, endG, endB, endW, endC, 1, 64, BIAS, 1, SB, TB, WHT);

  // segment softmax with end-bias
  k_fill<<<gdiv(N_BLOCKS), 256, 0, stream>>>(SEGMX, N_BLOCKS, -3.0e38f);
  k_segmax<<<gdiv(N_ATOMS), 256, 0, stream>>>(LOGITS, block_ids, SEGMX);
  k_m_eb<<<gdiv(N_BLOCKS), 256, 0, stream>>>(SEGMX, BIAS, MV, EBV, DEN);
  k_expsum<<<gdiv(N_ATOMS), 256, 0, stream>>>(LOGITS, MV, block_ids, DEN);
  k_write_p<<<gdiv((long)N_ATOMS * NLOGIT), 256, 0, stream>>>(LOGITS, DEN, block_ids, out);
  k_write_pend<<<gdiv(N_BLOCKS), 256, 0, stream>>>(EBV, DEN, out);
}
